// GCL_987842478182
// MI455X (gfx1250) — compile-verified, matrix operand order checked by reading, not run on hardware
//
#include <hip/hip_runtime.h>

#define D 128

typedef __attribute__((ext_vector_type(16))) _Float16 v16h;
typedef __attribute__((ext_vector_type(8)))  _Float16 v8h;
typedef __attribute__((ext_vector_type(4)))  _Float16 v4h;
typedef __attribute__((ext_vector_type(8)))  float    v8f;

// ---------------------------------------------------------------------------
// A-fragment load from LDS row (row-major, stride 256 halves).
// 16-bit A 16x32 layout: lane<16 holds K {0..7,16..23}, lane>=16 holds
// K {8..15,24..31} of its row -> two contiguous 16B DS loads per lane.
// kbase = kt*32 + (lane>>4)*8
// ---------------------------------------------------------------------------
__device__ __forceinline__ v16h load_a_frag(const _Float16* rowp, int kbase) {
  v8h lo = *(const v8h*)(rowp + kbase);
  v8h hi = *(const v8h*)(rowp + kbase + 16);
  v16h r;
#pragma unroll
  for (int j = 0; j < 8; ++j) { r[j] = lo[j]; r[j + 8] = hi[j]; }
  return r;
}

__device__ __forceinline__ float rsum16(float x) {
  x += __shfl_xor(x, 1, 32);
  x += __shfl_xor(x, 2, 32);
  x += __shfl_xor(x, 4, 32);
  x += __shfl_xor(x, 8, 32);
  return x;
}

// D = A[16 x 32*KT] * Wf + bias, A staged in LDS, Wf pre-swizzled fragments.
// acc[nt][i] holds element (row = i + 8*(lane>>4), col = nt*16 + (lane&15)).
template <int KT>
__device__ __forceinline__ void gemm_stage(_Float16 (*As)[256],
                                           const _Float16* __restrict__ Wf,
                                           const float* __restrict__ bias,
                                           int lane, v8f acc[8]) {
  const int cn  = lane & 15;
  const int hi8 = (lane >> 4) << 3;
  const int m   = lane & 15;
#pragma unroll
  for (int nt = 0; nt < 8; ++nt) {
    float bb = bias[nt * 16 + cn];
#pragma unroll
    for (int i = 0; i < 8; ++i) acc[nt][i] = bb;
  }
#pragma unroll
  for (int kt = 0; kt < KT; ++kt) {
    v16h a = load_a_frag(As[m], kt * 32 + hi8);
#pragma unroll
    for (int nt = 0; nt < 8; ++nt) {
      v16h b = *(const v16h*)(Wf + (((size_t)(kt * 8 + nt) * 32 + lane) << 4));
      acc[nt] = __builtin_amdgcn_wmma_f32_16x16x32_f16(
          /*neg_a=*/false, a, /*neg_b=*/false, b,
          /*c_mod=*/(short)0, acc[nt], /*reuse_a=*/false, /*reuse_b=*/false);
    }
  }
}

// Row-wise LayerNorm over 128 cols + SiLU, fully in registers via 16-lane
// xor-shuffle reductions (rows of a 16x16 C tile live in one 16-lane half).
__device__ __forceinline__ void ln_silu(v8f acc[8], const float* __restrict__ g,
                                        const float* __restrict__ b, int lane) {
  const int cn = lane & 15;
  float gv[8], bv[8];
#pragma unroll
  for (int nt = 0; nt < 8; ++nt) { gv[nt] = g[nt * 16 + cn]; bv[nt] = b[nt * 16 + cn]; }
#pragma unroll
  for (int i = 0; i < 8; ++i) {
    float s = 0.f, ss = 0.f;
#pragma unroll
    for (int nt = 0; nt < 8; ++nt) { float v = acc[nt][i]; s += v; ss += v * v; }
    s  = rsum16(s);
    ss = rsum16(ss);
    float mu  = s * (1.0f / 128.0f);
    float var = ss * (1.0f / 128.0f) - mu * mu;
    float inv = rsqrtf(var + 1e-5f);
#pragma unroll
    for (int nt = 0; nt < 8; ++nt) {
      float y = (acc[nt][i] - mu) * inv * gv[nt] + bv[nt];
      acc[nt][i] = y * (1.0f / (1.0f + __expf(-y)));   // SiLU
    }
  }
}

__device__ __forceinline__ void store_tile_f16(v8f acc[8], _Float16 (*As)[256], int lane) {
  const int cn    = lane & 15;
  const int rbase = (lane >> 4) << 3;
#pragma unroll
  for (int nt = 0; nt < 8; ++nt)
#pragma unroll
    for (int i = 0; i < 8; ++i)
      As[rbase + i][nt * 16 + cn] = (_Float16)acc[nt][i];
}

// ---------------------------------------------------------------------------
// Weight pre-swizzle: W[K x 128] f32 row-major -> per-(kt,nt,lane) v16h frags
// matching the 16-bit B 32x16 VGPR layout (mirror of A packing).
// ---------------------------------------------------------------------------
__global__ void pack_weights(const float* __restrict__ W, _Float16* __restrict__ out, int K) {
  int t = blockIdx.x * blockDim.x + threadIdx.x;
  int nKT = K >> 5;
  if (t >= nKT * 8 * 32) return;
  int lane = t & 31;
  int nt   = (t >> 5) & 7;
  int kt   = t >> 8;
  int n    = nt * 16 + (lane & 15);
  int hi   = (lane >> 4) << 3;
  _Float16* dst = out + ((size_t)t << 4);
#pragma unroll
  for (int j = 0; j < 16; ++j) {
    int k = kt * 32 + hi + (j & 7) + ((j >> 3) << 4);
    dst[j] = (_Float16)W[(size_t)k * D + n];
  }
}

// ---------------------------------------------------------------------------
// Edge kernel: one wave per 16-edge tile; 8 waves / block.
// ---------------------------------------------------------------------------
__global__ void __launch_bounds__(256, 1) egnn_edge(
    const float* __restrict__ h, const int* __restrict__ row, const int* __restrict__ col,
    const _Float16* __restrict__ We1f, const float* __restrict__ be1,
    const float* __restrict__ ge1, const float* __restrict__ bte1,
    const _Float16* __restrict__ We2f, const float* __restrict__ be2,
    const float* __restrict__ ge2, const float* __restrict__ bte2,
    float* __restrict__ mij, float* __restrict__ agg, int nE) {
  __shared__ _Float16 smem[8][16][256];
  const int lane = threadIdx.x & 31;
  const int wave = threadIdx.x >> 5;
  const int base = (blockIdx.x * 8 + wave) * 16;
  if (base >= nE) return;
  _Float16 (*As)[256] = smem[wave];

  // Gather concat(h[row], h[col]) -> LDS f16 [16][256]; 32 float4 per lane.
#pragma unroll 4
  for (int it = 0; it < 32; ++it) {
    int idx = it * 32 + lane;
    int e   = idx >> 6;          // edge in tile
    int q   = idx & 63;          // float4 slot in 256 floats
    int ei  = min(base + e, nE - 1);
    int src = (q < 32) ? row[ei] : col[ei];
    float4 v = ((const float4*)(h + (size_t)src * D))[q & 31];
    v4h pv = {(_Float16)v.x, (_Float16)v.y, (_Float16)v.z, (_Float16)v.w};
    *(v4h*)&As[e][q * 4] = pv;
  }
  asm volatile("s_wait_dscnt 0x0" ::: "memory");  // wave-local LDS phase fence

  v8f acc[8];
  gemm_stage<8>(As, We1f, be1, lane, acc);        // [16,256]@[256,128]
  ln_silu(acc, ge1, bte1, lane);
  store_tile_f16(acc, As, lane);
  asm volatile("s_wait_dscnt 0x0" ::: "memory");
  gemm_stage<4>(As, We2f, be2, lane, acc);        // [16,128]@[128,128]
  ln_silu(acc, ge2, bte2, lane);

  const int cn    = lane & 15;
  const int rbase = (lane >> 4) << 3;
#pragma unroll
  for (int i = 0; i < 8; ++i) {
    int e = base + rbase + i;
    if (e >= nE) continue;
    int r = row[e];
#pragma unroll
    for (int nt = 0; nt < 8; ++nt) {
      float v = acc[nt][i];
      mij[(size_t)e * D + nt * 16 + cn] = v;
      unsafeAtomicAdd(&agg[(size_t)r * D + nt * 16 + cn], v * 0.01f);  // /NORM_FACTOR
    }
  }
}

// ---------------------------------------------------------------------------
// Node kernel: one wave per 16-node tile.
// ---------------------------------------------------------------------------
__global__ void __launch_bounds__(256, 1) egnn_node(
    const float* __restrict__ h, const float* __restrict__ agg,
    const _Float16* __restrict__ Wn1f, const float* __restrict__ bn1,
    const float* __restrict__ gn1, const float* __restrict__ btn1,
    const _Float16* __restrict__ Wn2f, const float* __restrict__ bn2,
    float* __restrict__ hout, int nN) {
  __shared__ _Float16 smem[8][16][256];
  const int lane = threadIdx.x & 31;
  const int wave = threadIdx.x >> 5;
  const int base = (blockIdx.x * 8 + wave) * 16;
  if (base >= nN) return;
  _Float16 (*As)[256] = smem[wave];

#pragma unroll 4
  for (int it = 0; it < 32; ++it) {
    int idx = it * 32 + lane;
    int e   = idx >> 6;
    int q   = idx & 63;
    int ni  = min(base + e, nN - 1);
    const float* srcp = (q < 32) ? (h + (size_t)ni * D) : (agg + (size_t)ni * D);
    float4 v = ((const float4*)srcp)[q & 31];
    v4h pv = {(_Float16)v.x, (_Float16)v.y, (_Float16)v.z, (_Float16)v.w};
    *(v4h*)&As[e][q * 4] = pv;
  }
  asm volatile("s_wait_dscnt 0x0" ::: "memory");

  v8f acc[8];
  gemm_stage<8>(As, Wn1f, bn1, lane, acc);
  ln_silu(acc, gn1, btn1, lane);
  store_tile_f16(acc, As, lane);
  asm volatile("s_wait_dscnt 0x0" ::: "memory");
  gemm_stage<4>(As, Wn2f, bn2, lane, acc);        // no LN/SiLU after 2nd linear

  const int cn    = lane & 15;
  const int rbase = (lane >> 4) << 3;
#pragma unroll
  for (int i = 0; i < 8; ++i) {
    int n = base + rbase + i;
    if (n >= nN) continue;
#pragma unroll
    for (int nt = 0; nt < 8; ++nt) {
      size_t off = (size_t)n * D + nt * 16 + cn;
      hout[off] = h[off] + acc[nt][i];            // residual
    }
  }
}

// ---------------------------------------------------------------------------
extern "C" void kernel_launch(void* const* d_in, const int* in_sizes, int n_in,
                              void* d_out, int out_size, void* d_ws, size_t ws_size,
                              hipStream_t stream) {
  const float* h    = (const float*)d_in[0];
  const int*   row  = (const int*)d_in[1];
  const int*   col  = (const int*)d_in[2];
  const float* We1  = (const float*)d_in[3];
  const float* be1  = (const float*)d_in[4];
  const float* ge1  = (const float*)d_in[5];
  const float* bte1 = (const float*)d_in[6];
  const float* We2  = (const float*)d_in[7];
  const float* be2  = (const float*)d_in[8];
  const float* ge2  = (const float*)d_in[9];
  const float* bte2 = (const float*)d_in[10];
  const float* Wn1  = (const float*)d_in[11];
  const float* bn1  = (const float*)d_in[12];
  const float* gn1  = (const float*)d_in[13];
  const float* btn1 = (const float*)d_in[14];
  const float* Wn2  = (const float*)d_in[15];
  const float* bn2  = (const float*)d_in[16];

  const int nN = in_sizes[0] / D;
  const int nE = in_sizes[1];

  // Workspace layout: agg[N*D] f32, then four pre-swizzled f16 weight buffers.
  char* ws = (char*)d_ws;
  float* agg = (float*)ws;
  size_t off = ((size_t)nN * D * sizeof(float) + 255) & ~(size_t)255;
  _Float16* We1f = (_Float16*)(ws + off); off += (size_t)8 * 8 * 32 * 16 * sizeof(_Float16);
  _Float16* We2f = (_Float16*)(ws + off); off += (size_t)4 * 8 * 32 * 16 * sizeof(_Float16);
  _Float16* Wn1f = (_Float16*)(ws + off); off += (size_t)8 * 8 * 32 * 16 * sizeof(_Float16);
  _Float16* Wn2f = (_Float16*)(ws + off); off += (size_t)4 * 8 * 32 * 16 * sizeof(_Float16);

  hipMemsetAsync(agg, 0, (size_t)nN * D * sizeof(float), stream);
  pack_weights<<<(2048 + 255) / 256, 256, 0, stream>>>(We1, We1f, 256);
  pack_weights<<<(1024 + 255) / 256, 256, 0, stream>>>(We2, We2f, 128);
  pack_weights<<<(2048 + 255) / 256, 256, 0, stream>>>(Wn1, Wn1f, 256);
  pack_weights<<<(1024 + 255) / 256, 256, 0, stream>>>(Wn2, Wn2f, 128);

  float* hout = (float*)d_out;            // output order: (h_out, mij)
  float* mij  = hout + (size_t)nN * D;

  int etiles = (nE + 15) / 16;
  egnn_edge<<<(etiles + 7) / 8, 256, 0, stream>>>(h, row, col, We1f, be1, ge1, bte1,
                                                  We2f, be2, ge2, bte2, mij, agg, nE);
  int ntiles = (nN + 15) / 16;
  egnn_node<<<(ntiles + 7) / 8, 256, 0, stream>>>(h, agg, Wn1f, bn1, gn1, btn1,
                                                  Wn2f, bn2, hout, nN);
}
